// ProxemicsFieldGenerator_10668698763551
// MI455X (gfx1250) — matrix operand order repaired and checked by reading, not compiled, not run on hardware
//
#include <hip/hip_runtime.h>

typedef __attribute__((ext_vector_type(16))) _Float16 v16h;
typedef __attribute__((ext_vector_type(8)))  float    v8f;

#define N_NODES 1024
#define LRELU(x) ((x) >= 0.0f ? (x) : 0.2f * (x))

struct U4x2 { uint4 lo; uint4 hi; };

// ---------------- instance norm: one block per (b, c) ----------------
__global__ void instnorm_k(const float* __restrict__ x, float* __restrict__ y,
                           int n, int C) {
    int bc = blockIdx.x;
    int c = bc % C, b = bc / C;
    const float* xp = x + (size_t)b * n * C + c;
    float* yp = y + (size_t)b * n * C + c;
    int t = threadIdx.x;
    float s = 0.f, s2 = 0.f;
    for (int i = t; i < n; i += 256) {
        float v = xp[(size_t)i * C];
        s += v; s2 += v * v;
    }
    __shared__ float sh[256], sh2[256];
    sh[t] = s; sh2[t] = s2; __syncthreads();
    for (int o = 128; o > 0; o >>= 1) {
        if (t < o) { sh[t] += sh[t + o]; sh2[t] += sh2[t + o]; }
        __syncthreads();
    }
    float mean = sh[0] / (float)n;
    float var  = sh2[0] / (float)n - mean * mean;
    float r = __builtin_amdgcn_rsqf(var + 1e-5f);     // v_rsq_f32
    for (int i = t; i < n; i += 256)
        yp[(size_t)i * C] = (xp[(size_t)i * C] - mean) * r;
}

// ---------------- layer-0 projection: h' (f-major f16), s, d ----------------
__global__ void prep1_k(const float* __restrict__ xn, const float* __restrict__ w1,
                        const float* __restrict__ as1, const float* __restrict__ ad1,
                        _Float16* __restrict__ hpT, float* __restrict__ s,
                        float* __restrict__ d) {
    __shared__ float wl[4 * 32 * 16];
    __shared__ float asl[64], adl[64];
    int t = threadIdx.x;
    for (int i = t; i < 2048; i += 256) wl[i] = w1[i];
    if (t < 64) { asl[t] = as1[t]; adl[t] = ad1[t]; }
    __syncthreads();
    int gid = blockIdx.x * 256 + t;          // b*1024 + n
    int n = gid & 1023, b = gid >> 10;
    float xr[32];
    const float* xp = xn + (size_t)gid * 32;
#pragma unroll
    for (int f = 0; f < 32; ++f) xr[f] = xp[f];
    for (int h = 0; h < 4; ++h) {
        float ss = 0.f, dd = 0.f;
        for (int o = 0; o < 16; ++o) {
            float acc = 0.f;
#pragma unroll
            for (int f = 0; f < 32; ++f) acc += xr[f] * wl[(h * 32 + f) * 16 + o];
            hpT[((((size_t)b * 4 + h) * 16 + o) * N_NODES) + n] = (_Float16)acc;
            ss += acc * asl[h * 16 + o];
            dd += acc * adl[h * 16 + o];
        }
        s[((size_t)b * 4 + h) * N_NODES + n] = ss;
        d[((size_t)b * 4 + h) * N_NODES + n] = dd;
    }
}

// ---------------- layer-1 projection ----------------
__global__ void prep2_k(const float* __restrict__ xn, const float* __restrict__ w2,
                        const float* __restrict__ as2, const float* __restrict__ ad2,
                        _Float16* __restrict__ hpT, float* __restrict__ s,
                        float* __restrict__ d) {
    __shared__ float wl[64 * 32];
    __shared__ float asl[32], adl[32];
    int t = threadIdx.x;
    for (int i = t; i < 2048; i += 256) wl[i] = w2[i];
    if (t < 32) { asl[t] = as2[t]; adl[t] = ad2[t]; }
    __syncthreads();
    int gid = blockIdx.x * 256 + t;
    int n = gid & 1023, b = gid >> 10;
    float xr[64];
    const float* xp = xn + (size_t)gid * 64;
#pragma unroll
    for (int f = 0; f < 64; ++f) xr[f] = xp[f];
    float ss = 0.f, dd = 0.f;
    for (int o = 0; o < 32; ++o) {
        float acc = 0.f;
#pragma unroll
        for (int f = 0; f < 64; ++f) acc += xr[f] * wl[f * 32 + o];
        hpT[((size_t)b * 32 + o) * N_NODES + n] = (_Float16)acc;
        ss += acc * asl[o];
        dd += acc * adl[o];
    }
    s[(size_t)b * N_NODES + n] = ss;
    d[(size_t)b * N_NODES + n] = dd;
}

// ---------------- dmax reduce: one block per (b*h) group ----------------
__global__ void dmax_k(const float* __restrict__ d, float* __restrict__ dmax, int n) {
    int g = blockIdx.x, t = threadIdx.x;
    const float* dp = d + (size_t)g * n;
    float mx = -3.4e38f;
    for (int i = t; i < n; i += 256) mx = fmaxf(mx, dp[i]);
    __shared__ float sh[256];
    sh[t] = mx; __syncthreads();
    for (int o = 128; o > 0; o >>= 1) {
        if (t < o) sh[t] = fmaxf(sh[t], sh[t + o]);
        __syncthreads();
    }
    if (t == 0) dmax[g] = sh[0];
}

// A-fragment K index, contiguous order k = 16*khalf + s (pairs per VGPR)
__device__ __forceinline__ int a_klocal(int v, int khalf) {
    return 2 * v + khalf * 16;
}

// ---------------- fused attention, layer 0 (h0=4, f1=16) ----------------
// 4 waves per block, each wave owns one 16-row tile of the same (b,h);
// shared d-row staged once. Writes ELU'd [b,n,64] mid tensor.
__global__ __launch_bounds__(128) void attn1_k(
    const _Float16* __restrict__ hpT, const float* __restrict__ s,
    const float* __restrict__ d, const float* __restrict__ dmax,
    const float* __restrict__ bias, float* __restrict__ xmid) {
    int blk = blockIdx.x;                    // bh * 16 + tilegrp
    int tilegrp = blk & 15, bh = blk >> 4;
    int tid = threadIdx.x;
    int wave = tid >> 5, lane = tid & 31;
    int tile = tilegrp * 4 + wave;
    int m = lane & 15, khalf = lane >> 4;

    __shared__ float dsh[N_NODES];
    const float* drow = d + (size_t)bh * N_NODES;
    for (int i = tid; i < N_NODES; i += 128) dsh[i] = drow[i];
    __syncthreads();

    float sm = s[(size_t)bh * N_NODES + tile * 16 + m];
    float t0 = sm + dmax[bh];
    float mrow = LRELU(t0);

    v8f c = {};
    float lsum = 0.f;
    const _Float16* bp = hpT + (size_t)bh * 16 * N_NODES;

    for (int jc = 0; jc < N_NODES; jc += 32) {
        v16h afrag;
#pragma unroll
        for (int v = 0; v < 8; ++v) {
            int kl = a_klocal(v, khalf);
            float sc0 = sm + dsh[jc + kl];
            float sc1 = sm + dsh[jc + kl + 1];
            float p0 = __expf(LRELU(sc0) - mrow);
            float p1 = __expf(LRELU(sc1) - mrow);
            lsum += p0 + p1;
            afrag[2 * v]     = (_Float16)p0;
            afrag[2 * v + 1] = (_Float16)p1;
        }
        const _Float16* col = bp + (size_t)m * N_NODES + jc + khalf * 16;
        U4x2 raw;
        raw.lo = *(const uint4*)(col);
        raw.hi = *(const uint4*)(col + 8);
        v16h bfrag = __builtin_bit_cast(v16h, raw);
        c = __builtin_amdgcn_wmma_f32_16x16x32_f16(false, afrag, false, bfrag,
                                                   (short)0, c, false, false);
    }

    __shared__ float lsh[128];
    lsh[tid] = lsum;
    __syncthreads();
    const float* lw = lsh + wave * 32;

    int h = bh & 3, b = bh >> 2;
    float bv = bias[m];
    float* orow = xmid + ((size_t)(b * N_NODES + tile * 16)) * 64 + h * 16 + m;
#pragma unroll
    for (int r = 0; r < 8; ++r) {
        int M = r + 8 * khalf;
        float inv = __builtin_amdgcn_rcpf(lw[M] + lw[M + 16]);   // v_rcp_f32
        float val = c[r] * inv + bv;
        float e = val > 0.f ? val : __expf(val) - 1.0f;          // ELU
        orow[(size_t)M * 64] = e;
    }
}

// ---------------- fused attention, layer 1 (h=1, f2=32) ----------------
__global__ __launch_bounds__(128) void attn2_k(
    const _Float16* __restrict__ hpT, const float* __restrict__ s,
    const float* __restrict__ d, const float* __restrict__ dmax,
    const float* __restrict__ bias, float* __restrict__ out) {
    int blk = blockIdx.x;                    // b * 16 + tilegrp
    int tilegrp = blk & 15, b = blk >> 4;
    int tid = threadIdx.x;
    int wave = tid >> 5, lane = tid & 31;
    int tile = tilegrp * 4 + wave;
    int m = lane & 15, khalf = lane >> 4;

    __shared__ float dsh[N_NODES];
    const float* drow = d + (size_t)b * N_NODES;
    for (int i = tid; i < N_NODES; i += 128) dsh[i] = drow[i];
    __syncthreads();

    float sm = s[(size_t)b * N_NODES + tile * 16 + m];
    float t0 = sm + dmax[b];
    float mrow = LRELU(t0);

    v8f c0 = {}, c1 = {};
    float lsum = 0.f;
    const _Float16* bp = hpT + (size_t)b * 32 * N_NODES;

    for (int jc = 0; jc < N_NODES; jc += 32) {
        v16h afrag;
#pragma unroll
        for (int v = 0; v < 8; ++v) {
            int kl = a_klocal(v, khalf);
            float sc0 = sm + dsh[jc + kl];
            float sc1 = sm + dsh[jc + kl + 1];
            float p0 = __expf(LRELU(sc0) - mrow);
            float p1 = __expf(LRELU(sc1) - mrow);
            lsum += p0 + p1;
            afrag[2 * v]     = (_Float16)p0;
            afrag[2 * v + 1] = (_Float16)p1;
        }
        const _Float16* colA = bp + (size_t)m * N_NODES + jc + khalf * 16;
        const _Float16* colB = bp + (size_t)(16 + m) * N_NODES + jc + khalf * 16;
        U4x2 rawA, rawB;
        rawA.lo = *(const uint4*)(colA); rawA.hi = *(const uint4*)(colA + 8);
        rawB.lo = *(const uint4*)(colB); rawB.hi = *(const uint4*)(colB + 8);
        v16h bfragA = __builtin_bit_cast(v16h, rawA);
        v16h bfragB = __builtin_bit_cast(v16h, rawB);
        c0 = __builtin_amdgcn_wmma_f32_16x16x32_f16(false, afrag, false, bfragA,
                                                    (short)0, c0, false, false);
        c1 = __builtin_amdgcn_wmma_f32_16x16x32_f16(false, afrag, false, bfragB,
                                                    (short)0, c1, false, false);
    }

    __shared__ float lsh[128];
    lsh[tid] = lsum;
    __syncthreads();
    const float* lw = lsh + wave * 32;

    float bv0 = bias[m], bv1 = bias[16 + m];
    float* orow = out + ((size_t)(b * N_NODES + tile * 16)) * 32;
#pragma unroll
    for (int r = 0; r < 8; ++r) {
        int M = r + 8 * khalf;
        float inv = __builtin_amdgcn_rcpf(lw[M] + lw[M + 16]);
        orow[(size_t)M * 32 + m]      = c0[r] * inv + bv0;
        orow[(size_t)M * 32 + 16 + m] = c1[r] * inv + bv1;
    }
}

static inline size_t align256(size_t x) { return (x + 255) & ~(size_t)255; }

extern "C" void kernel_launch(void* const* d_in, const int* in_sizes, int n_in,
                              void* d_out, int out_size, void* d_ws, size_t ws_size,
                              hipStream_t stream) {
    const float* x      = (const float*)d_in[0];   // [16,1024,32]
    const float* w1     = (const float*)d_in[1];   // [4,32,16]
    const float* a_src1 = (const float*)d_in[2];   // [4,16,1]
    const float* a_dst1 = (const float*)d_in[3];
    const float* b1     = (const float*)d_in[4];   // [16]
    const float* w2     = (const float*)d_in[5];   // [1,64,32]
    const float* a_src2 = (const float*)d_in[6];   // [1,32,1]
    const float* a_dst2 = (const float*)d_in[7];
    const float* b2     = (const float*)d_in[8];   // [32]
    float* out = (float*)d_out;                    // [16,1024,32]

    const int B = 16, N = N_NODES;
    char* ws = (char*)d_ws;
    size_t off = 0;
    float* xn1 = (float*)(ws + off);            off += align256((size_t)B * N * 32 * 4);
    _Float16* hpT1 = (_Float16*)(ws + off);     off += align256((size_t)B * 4 * 16 * N * 2);
    float* s1 = (float*)(ws + off);             off += align256((size_t)B * 4 * N * 4);
    float* d1 = (float*)(ws + off);             off += align256((size_t)B * 4 * N * 4);
    float* dmax1 = (float*)(ws + off);          off += 256;
    float* xmid = (float*)(ws + off);           off += align256((size_t)B * N * 64 * 4);
    float* xn2 = (float*)(ws + off);            off += align256((size_t)B * N * 64 * 4);
    _Float16* hpT2 = (_Float16*)(ws + off);     off += align256((size_t)B * 32 * N * 2);
    float* s2 = (float*)(ws + off);             off += align256((size_t)B * N * 4);
    float* d2 = (float*)(ws + off);             off += align256((size_t)B * N * 4);
    float* dmax2 = (float*)(ws + off);          off += 256;

    // layer 0
    instnorm_k<<<B * 32, 256, 0, stream>>>(x, xn1, N, 32);
    prep1_k<<<B * N / 256, 256, 0, stream>>>(xn1, w1, a_src1, a_dst1, hpT1, s1, d1);
    dmax_k<<<B * 4, 256, 0, stream>>>(d1, dmax1, N);
    attn1_k<<<B * 4 * (N / 64), 128, 0, stream>>>(hpT1, s1, d1, dmax1, b1, xmid);
    // layer 1
    instnorm_k<<<B * 64, 256, 0, stream>>>(xmid, xn2, N, 64);
    prep2_k<<<B * N / 256, 256, 0, stream>>>(xn2, w2, a_src2, a_dst2, hpT2, s2, d2);
    dmax_k<<<B, 256, 0, stream>>>(d2, dmax2, N);
    attn2_k<<<B * (N / 64), 128, 0, stream>>>(hpT2, s2, d2, dmax2, b2, out);
}
